// FinanceGPT_77386720739530
// MI455X (gfx1250) — matrix operand order repaired and checked by reading, not compiled, not run on hardware
//
#include <hip/hip_runtime.h>

typedef __bf16 bf16;
typedef __bf16 v16bf __attribute__((ext_vector_type(16)));
typedef float  v8f   __attribute__((ext_vector_type(8)));
typedef unsigned int u32x4 __attribute__((ext_vector_type(4)));
typedef int          i32x4 __attribute__((ext_vector_type(4)));
typedef int          i32x8 __attribute__((ext_vector_type(8)));

union FragB16 { uint4 u[2]; v16bf v; };

__device__ __forceinline__ v8f wmma_bf16(v16bf a, v16bf b, v8f c) {
  // 8 args: (neg_a, A, neg_b, B, c_mod, C, reuse_a, reuse_b)
  return __builtin_amdgcn_wmma_f32_16x16x32_bf16(false, a, false, b, (short)0, c, false, false);
}

// ---------------------------------------------------------------------------
// TDM: async 2D tile load global -> LDS (D# per CDNA5 ISA)
//   dslog: data_size code (1 = 2 bytes). tile0 elems contiguous, tile1 rows,
//   stride0 = row stride in elements. Issued once per wave; tracked by TENSORcnt.
// ---------------------------------------------------------------------------
__device__ __forceinline__ void tdm_load_2d(unsigned lds_off, const void* gptr,
                                            unsigned dim0, unsigned dim1,
                                            unsigned stride0,
                                            unsigned tile0, unsigned tile1,
                                            unsigned dslog) {
  const unsigned long long ga = (unsigned long long)gptr;
  u32x4 g0;
  g0.x = 1u;                                               // count=1 (valid)
  g0.y = lds_off;                                          // lds_addr (bytes)
  g0.z = (unsigned)ga;                                     // global_addr[31:0]
  g0.w = (unsigned)((ga >> 32) & 0x1FFFFFFu) | (2u << 30); // addr[56:32] | type=2
  i32x8 g1;
  g1[0] = (int)(dslog << 16);                              // data_size, mask=0
  g1[1] = (int)((dim0 & 0xFFFFu) << 16);                   // tensor_dim0[15:0]
  g1[2] = (int)((dim0 >> 16) | ((dim1 & 0xFFFFu) << 16));  // td0 hi | td1 lo
  g1[3] = (int)((dim1 >> 16) | (tile0 << 16));             // td1 hi | tile_dim0
  g1[4] = (int)(tile1 & 0xFFFFu);                          // tile_dim1, tile_dim2=0
  g1[5] = (int)stride0;                                    // tensor_dim0_stride[31:0]
  g1[6] = 0;
  g1[7] = 0;
  const i32x4 z4 = {0, 0, 0, 0};
#if __clang_major__ >= 23
  const i32x8 z8 = {0, 0, 0, 0, 0, 0, 0, 0};
  __builtin_amdgcn_tensor_load_to_lds(g0, g1, z4, z4, z8, 0);
#else
  __builtin_amdgcn_tensor_load_to_lds(g0, g1, z4, z4, 0);
#endif
}

// ---------------------------------------------------------------------------
// Embedding gather: x[row,:] = embed[idx[row],:]   (row = b*T + t, D = 1024)
// ---------------------------------------------------------------------------
__global__ __launch_bounds__(256) void gather_k(const int* __restrict__ idx,
                                                const float* __restrict__ embed,
                                                float* __restrict__ x) {
  const int row = blockIdx.x;
  const int tok = idx[row];
  const float4 e = *(const float4*)(embed + (size_t)tok * 1024 + threadIdx.x * 4);
  *(float4*)(x + (size_t)row * 1024 + threadIdx.x * 4) = e;
}

// ---------------------------------------------------------------------------
// RMSNorm (D=1024) fp32 in -> bf16 out, one row per 256-thread block
// ---------------------------------------------------------------------------
__global__ __launch_bounds__(256) void rmsnorm_k(const float* __restrict__ x,
                                                 const float* __restrict__ w,
                                                 bf16* __restrict__ out) {
  __shared__ float red[256];
  const int row = blockIdx.x;
  const int tid = threadIdx.x;
  const float4 xv = *(const float4*)(x + (size_t)row * 1024 + tid * 4);
  red[tid] = xv.x * xv.x + xv.y * xv.y + xv.z * xv.z + xv.w * xv.w;
  __syncthreads();
  for (int off = 128; off > 0; off >>= 1) {
    if (tid < off) red[tid] += red[tid + off];
    __syncthreads();
  }
  const float scale = rsqrtf(red[0] * (1.0f / 1024.0f) + 1e-6f);
  const float4 wv = *(const float4*)(w + tid * 4);
  bf16* o = out + (size_t)row * 1024 + tid * 4;
  o[0] = (bf16)(xv.x * scale * wv.x);
  o[1] = (bf16)(xv.y * scale * wv.y);
  o[2] = (bf16)(xv.z * scale * wv.z);
  o[3] = (bf16)(xv.w * scale * wv.w);
}

// ---------------------------------------------------------------------------
// Tiled bf16 WMMA GEMM: C[2048, N] = A[2048, K](bf16) @ B(fp32)
//   BT=false: B is [K,N] row-major.  BT=true: B is [N,K] row-major (embed^T).
//   OUTB: store bf16, else fp32.  RES: C = Res + A@B (fp32 out).
// Block tile 128x64, 8 waves (4x2), wave tile 32x32 = 2x2 WMMA 16x16x32.
// A tile staged by TDM (double-buffered, wave 0 issues, TENSORcnt pipelined);
// B tile staged manually (fp32 -> bf16 transpose).
// ---------------------------------------------------------------------------
template <bool BT, bool OUTB, bool RES>
__global__ __launch_bounds__(256) void gemm_k(const bf16* __restrict__ A,
                                              const float* __restrict__ Bw,
                                              void* __restrict__ Cout,
                                              const float* __restrict__ Res,
                                              int Ndim, int Kdim) {
  __shared__ bf16 As[2][128 * 32];  // TDM destination, row stride 32 (64 B)
  __shared__ bf16 Bt[64 * 40];      // n-major (transposed), stride 40

  const int tid  = threadIdx.x;
  const int lane = tid & 31;
  const int wid  = tid >> 5;
  const int wm   = (wid & 3) * 32;
  const int wn   = (wid >> 2) * 32;
  const int n0   = blockIdx.x * 64;
  const int m0   = blockIdx.y * 128;

  v8f acc[2][2] = {};

  const int hi    = (lane >> 4);       // 0 or 1
  const int khalf = hi * 8;            // A-frag K group base
  const int kb2   = hi * 8;            // B-frag K base
  const int col   = lane & 15;

  const int niter = Kdim >> 5;
  // preload A tile 0 via TDM (tile = 32 contiguous k x 128 rows, bf16)
  if (wid == 0)
    tdm_load_2d((unsigned)(size_t)&As[0][0], A + (size_t)m0 * Kdim,
                (unsigned)Kdim, 0x100000u, (unsigned)Kdim, 32u, 128u, 1u);

  for (int it = 0; it < niter; ++it) {
    const int k0 = it << 5;
    const bool more = (it + 1) < niter;
    // issue TDM for next A tile into the other buffer
    if (wid == 0 && more)
      tdm_load_2d((unsigned)(size_t)&As[(it + 1) & 1][0],
                  A + (size_t)m0 * Kdim + (k0 + 32),
                  (unsigned)Kdim, 0x100000u, (unsigned)Kdim, 32u, 128u, 1u);

    // ---- stage B tile -> Bt[n][k] (fp32 -> bf16, transposed) ----
    if (!BT) {
#pragma unroll
      for (int i = 0; i < 2; ++i) {
        const int e  = (tid + i * 256) * 4;   // 0..2047
        const int kk = e >> 6;
        const int nn = e & 63;
        const float4 f = *(const float4*)(Bw + (size_t)(k0 + kk) * Ndim + n0 + nn);
        Bt[(nn + 0) * 40 + kk] = (bf16)f.x;
        Bt[(nn + 1) * 40 + kk] = (bf16)f.y;
        Bt[(nn + 2) * 40 + kk] = (bf16)f.z;
        Bt[(nn + 3) * 40 + kk] = (bf16)f.w;
        if (more) __builtin_prefetch(Bw + (size_t)(k0 + 32 + kk) * Ndim + n0 + nn, 0, 1);
      }
    } else {
#pragma unroll
      for (int i = 0; i < 2; ++i) {
        const int e  = (tid + i * 256) * 4;
        const int nn = e >> 5;
        const int kk = e & 31;
        const float4 f = *(const float4*)(Bw + (size_t)(n0 + nn) * Kdim + k0 + kk);
        bf16* d = &Bt[nn * 40 + kk];
        d[0] = (bf16)f.x; d[1] = (bf16)f.y; d[2] = (bf16)f.z; d[3] = (bf16)f.w;
        if (more) __builtin_prefetch(Bw + (size_t)(n0 + nn) * Kdim + k0 + 32 + kk, 0, 1);
      }
    }

    // current A tile (issued last iteration) must be complete; TDM ops from one
    // wave complete in order, so <=1 outstanding means the older one is done.
    if (more) __builtin_amdgcn_s_wait_tensorcnt(1);
    else      __builtin_amdgcn_s_wait_tensorcnt(0);
    __syncthreads();

    // ---- fragments ----
    const bf16* Acur = &As[it & 1][0];
    FragB16 af[2], bfr[2];
#pragma unroll
    for (int mi = 0; mi < 2; ++mi) {
      const bf16* p = Acur + (wm + mi * 16 + col) * 32 + khalf;
      af[mi].u[0] = *(const uint4*)(p);        // K khalf..khalf+7
      af[mi].u[1] = *(const uint4*)(p + 16);   // K khalf+16..khalf+23
    }
#pragma unroll
    for (int ni = 0; ni < 2; ++ni) {
      const bf16* p = &Bt[(wn + ni * 16 + col) * 40 + kb2];
      bfr[ni].u[0] = *(const uint4*)(p);       // K kb2..kb2+7
      bfr[ni].u[1] = *(const uint4*)(p + 16);  // K kb2+16..kb2+23
    }
#pragma unroll
    for (int mi = 0; mi < 2; ++mi)
#pragma unroll
      for (int ni = 0; ni < 2; ++ni)
        acc[mi][ni] = wmma_bf16(af[mi].v, bfr[ni].v, acc[mi][ni]);
    __syncthreads();
  }

  // ---- epilogue ----
  const int rbase = hi * 8;
#pragma unroll
  for (int mi = 0; mi < 2; ++mi)
#pragma unroll
    for (int ni = 0; ni < 2; ++ni)
#pragma unroll
      for (int r = 0; r < 8; ++r) {
        const int gm = m0 + wm + mi * 16 + rbase + r;
        const int gn = n0 + wn + ni * 16 + col;
        float v = acc[mi][ni][r];
        const size_t off = (size_t)gm * Ndim + gn;
        if (RES) v += Res[off];
        if (OUTB) ((bf16*)Cout)[off] = (bf16)v;
        else      ((float*)Cout)[off] = v;
      }
}

// ---------------------------------------------------------------------------
// RoPE + head split: qkv fp32 [2048, 3072] ->
//   qh/kh bf16 [B,H,T,64] with rope applied; vth bf16 [B,H,64,T] (transposed)
// ---------------------------------------------------------------------------
__global__ __launch_bounds__(256) void rope_k(const float* __restrict__ qkv,
                                              bf16* __restrict__ qh,
                                              bf16* __restrict__ kh,
                                              bf16* __restrict__ vth) {
  const int r = blockIdx.x;            // 0..2047
  const int bb = r >> 10;
  const int t  = r & 1023;
  const float* row = qkv + (size_t)r * 3072;
  const int tid = threadIdx.x;

#pragma unroll
  for (int pp = 0; pp < 2; ++pp) {
    const int p = tid + pp * 256;      // 0..511 : (head, pair)
    const int h = p >> 5;
    const int i = p & 31;
    // inv_freq = 10000^(-2i/64) = exp(-i * ln(10000)/32)
    const float freq = (float)t * __expf(-0.28782313662425572f * (float)i);
    const float c = __cosf(freq), s = __sinf(freq);
    const int base = h * 64 + 2 * i;
    const float qa = row[base], qb2 = row[base + 1];
    const float ka = row[1024 + base], kb2 = row[1024 + base + 1];
    const size_t o = ((size_t)(bb * 16 + h) * 1024 + t) * 64 + 2 * i;
    qh[o]     = (bf16)(qa * c - qb2 * s);
    qh[o + 1] = (bf16)(qa * s + qb2 * c);
    kh[o]     = (bf16)(ka * c - kb2 * s);
    kh[o + 1] = (bf16)(ka * s + kb2 * c);
  }
#pragma unroll
  for (int jj = 0; jj < 4; ++jj) {
    const int j = tid + jj * 256;      // 0..1023
    const int h = j >> 6;
    const int d = j & 63;
    vth[((size_t)(bb * 16 + h) * 64 + d) * 1024 + t] = (bf16)row[2048 + j];
  }
}

// ---------------------------------------------------------------------------
// Flash attention, one wave per (b, h, 16-row q tile).  d_k = 64, T = 1024.
// All K fragments (and all V fragments) are issued before their WMMAs so the
// b128 loads pipeline instead of load->wait->wmma serialization.
// ---------------------------------------------------------------------------
__global__ __launch_bounds__(32) void attn_k(const bf16* __restrict__ qh,
                                             const bf16* __restrict__ kh,
                                             const bf16* __restrict__ vth,
                                             bf16* __restrict__ yb) {
  __shared__ float S[16 * 32];
  __shared__ bf16  P[16 * 32];
  __shared__ float sm[16], sl[16], sa[16];

  const int lane = threadIdx.x;
  const int qt = blockIdx.x & 63;
  const int bh = blockIdx.x >> 6;
  const int q0 = qt * 16;
  const int bb = bh >> 4;
  const int h  = bh & 15;

  const bf16* Qb = qh + ((size_t)bh * 1024 + q0) * 64;
  const bf16* Kb = kh + (size_t)bh * 1024 * 64;
  const bf16* Vb = vth + (size_t)bh * 64 * 1024;

  const int mrow  = lane & 15;
  const int hi    = lane >> 4;
  const int khalf = hi * 8;
  const int rbase = hi * 8;

  // Q fragments: 16x64 split into two 16x32 A-fragments
  FragB16 qa[2];
#pragma unroll
  for (int c = 0; c < 2; ++c) {
    const bf16* p = Qb + mrow * 64 + c * 32 + khalf;
    qa[c].u[0] = *(const uint4*)(p);
    qa[c].u[1] = *(const uint4*)(p + 16);
  }

  if (lane < 16) { sm[lane] = -1e30f; sl[lane] = 0.0f; }
  __syncthreads();

  v8f oacc[4] = {};
  const int nch = (q0 + 16 + 31) >> 5;
  for (int kc = 0; kc < nch; ++kc) {
    const int k0 = kc * 32;
    // ---- S(16x32) = Q(16x64) @ K^T(64x32): load all 4 B-frags, then 4 WMMAs
    FragB16 kf[2][2];  // [c][nsub]
#pragma unroll
    for (int c = 0; c < 2; ++c)
#pragma unroll
      for (int ns = 0; ns < 2; ++ns) {
        const bf16* p = Kb + (size_t)(k0 + ns * 16 + mrow) * 64 + c * 32 + hi * 8;
        kf[c][ns].u[0] = *(const uint4*)(p);
        kf[c][ns].u[1] = *(const uint4*)(p + 16);
      }
    if (kc + 1 < nch)
      __builtin_prefetch(Kb + (size_t)(k0 + 32 + mrow) * 64, 0, 1);
    v8f s0 = {}, s1 = {};
    s0 = wmma_bf16(qa[0].v, kf[0][0].v, s0);
    s1 = wmma_bf16(qa[0].v, kf[0][1].v, s1);
    s0 = wmma_bf16(qa[1].v, kf[1][0].v, s0);
    s1 = wmma_bf16(qa[1].v, kf[1][1].v, s1);
#pragma unroll
    for (int r = 0; r < 8; ++r) {
      S[(rbase + r) * 32 + mrow]      = s0[r] * 0.125f;   // 1/sqrt(64)
      S[(rbase + r) * 32 + 16 + mrow] = s1[r] * 0.125f;
    }
    __syncthreads();

    // ---- streaming softmax stats (16 threads, one row each) ----
    if (lane < 16) {
      const int q = q0 + lane;
      const float mo = sm[lane];
      float mx = mo;
      for (int j = 0; j < 32; ++j) {
        const float v = (k0 + j <= q) ? S[lane * 32 + j] : -1e30f;
        mx = fmaxf(mx, v);
      }
      const float al = __expf(mo - mx);
      float ss = 0.0f;
      for (int j = 0; j < 32; ++j) {
        const float pv = (k0 + j <= q) ? __expf(S[lane * 32 + j] - mx) : 0.0f;
        P[lane * 32 + j] = (bf16)pv;
        ss += pv;
      }
      sl[lane] = sl[lane] * al + ss;
      sm[lane] = mx;
      sa[lane] = al;
    }
    __syncthreads();

    // ---- rescale O by alpha[row]; load P frag + all 4 V frags, then 4 WMMAs
    FragB16 vf[4];
#pragma unroll
    for (int t4 = 0; t4 < 4; ++t4) {
      const bf16* vp = Vb + (size_t)(t4 * 16 + mrow) * 1024 + k0 + hi * 8;
      vf[t4].u[0] = *(const uint4*)(vp);
      vf[t4].u[1] = *(const uint4*)(vp + 16);
    }
    float alr[8];
#pragma unroll
    for (int r = 0; r < 8; ++r) alr[r] = sa[rbase + r];
#pragma unroll
    for (int t4 = 0; t4 < 4; ++t4)
#pragma unroll
      for (int r = 0; r < 8; ++r) oacc[t4][r] *= alr[r];

    FragB16 pf;
    const bf16* pp = &P[mrow * 32 + khalf];
    pf.u[0] = *(const uint4*)(pp);
    pf.u[1] = *(const uint4*)(pp + 16);
#pragma unroll
    for (int t4 = 0; t4 < 4; ++t4)
      oacc[t4] = wmma_bf16(pf.v, vf[t4].v, oacc[t4]);
    __syncthreads();
  }

  // ---- normalize and store ----
  float lr[8];
#pragma unroll
  for (int r = 0; r < 8; ++r) lr[r] = 1.0f / sl[rbase + r];
#pragma unroll
  for (int t4 = 0; t4 < 4; ++t4)
#pragma unroll
    for (int r = 0; r < 8; ++r) {
      const size_t orow = (size_t)bb * 1024 + q0 + rbase + r;
      yb[orow * 1024 + h * 64 + t4 * 16 + mrow] = (bf16)(oacc[t4][r] * lr[r]);
    }
}

// ---------------------------------------------------------------------------
// g1 = silu(g1) * g2   (bf16, 2048*4096 elems, 4 per thread)
// ---------------------------------------------------------------------------
__global__ __launch_bounds__(256) void silu_mul_k(bf16* __restrict__ g1,
                                                  const bf16* __restrict__ g2) {
  const size_t i = ((size_t)blockIdx.x * 256 + threadIdx.x) * 4;
#pragma unroll
  for (int j = 0; j < 4; ++j) {
    const float a = (float)g1[i + j];
    const float b = (float)g2[i + j];
    g1[i + j] = (bf16)(a * b / (1.0f + __expf(-a)));
  }
}

// ---------------------------------------------------------------------------
extern "C" void kernel_launch(void* const* d_in, const int* in_sizes, int n_in,
                              void* d_out, int out_size, void* d_ws, size_t ws_size,
                              hipStream_t stream) {
  (void)in_sizes; (void)n_in; (void)out_size; (void)ws_size;
  const int*   idx   = (const int*)d_in[0];
  const float* embed = (const float*)d_in[1];
  const float* ln1   = (const float*)d_in[2];
  const float* wqkv  = (const float*)d_in[3];
  const float* wo    = (const float*)d_in[4];
  const float* ln2   = (const float*)d_in[5];
  const float* wg    = (const float*)d_in[6];
  const float* wu    = (const float*)d_in[7];
  const float* wd    = (const float*)d_in[8];
  const float* lnf   = (const float*)d_in[9];
  float* out = (float*)d_out;

  char* ws = (char*)d_ws;
  float* x   = (float*)(ws + 0);          //  8 MB fp32 [2048,1024]
  bf16*  hb  = (bf16*) (ws + 8388608);    //  4 MB bf16 [2048,1024]
  float* qkv = (float*)(ws + 12582912);   // 24 MB fp32 [2048,3072]
  bf16*  qh  = (bf16*) (ws + 37748736);   //  4 MB bf16 [B,H,T,64]
  bf16*  kh  = (bf16*) (ws + 41943040);   //  4 MB bf16 [B,H,T,64]
  bf16*  vth = (bf16*) (ws + 46137344);   //  4 MB bf16 [B,H,64,T]
  bf16*  yb  = (bf16*) (ws + 50331648);   //  4 MB bf16 [2048,1024]
  bf16*  g1  = (bf16*) (ws + 54525952);   // 16 MB bf16 [2048,4096]
  bf16*  g2  = (bf16*) (ws + 71303168);   // 16 MB bf16 [2048,4096]

  gather_k<<<2048, 256, 0, stream>>>(idx, embed, x);

  for (int l = 0; l < 6; ++l) {
    rmsnorm_k<<<2048, 256, 0, stream>>>(x, ln1 + (size_t)l * 1024, hb);
    gemm_k<false, false, false><<<dim3(3072 / 64, 16), 256, 0, stream>>>(
        hb, wqkv + (size_t)l * 1024 * 3072, qkv, nullptr, 3072, 1024);
    rope_k<<<2048, 256, 0, stream>>>(qkv, qh, kh, vth);
    attn_k<<<2048, 32, 0, stream>>>(qh, kh, vth, yb);
    gemm_k<false, false, true><<<dim3(1024 / 64, 16), 256, 0, stream>>>(
        yb, wo + (size_t)l * 1024 * 1024, x, x, 1024, 1024);
    rmsnorm_k<<<2048, 256, 0, stream>>>(x, ln2 + (size_t)l * 1024, hb);
    gemm_k<false, true, false><<<dim3(4096 / 64, 16), 256, 0, stream>>>(
        hb, wg + (size_t)l * 1024 * 4096, g1, nullptr, 4096, 1024);
    gemm_k<false, true, false><<<dim3(4096 / 64, 16), 256, 0, stream>>>(
        hb, wu + (size_t)l * 1024 * 4096, g2, nullptr, 4096, 1024);
    silu_mul_k<<<8192, 256, 0, stream>>>(g1, g2);
    gemm_k<false, false, true><<<dim3(1024 / 64, 16), 256, 0, stream>>>(
        g1, wd + (size_t)l * 4096 * 1024, x, x, 1024, 4096);
  }

  rmsnorm_k<<<2048, 256, 0, stream>>>(x, lnf, hb);
  gemm_k<true, false, false><<<dim3(32000 / 64, 16), 256, 0, stream>>>(
      hb, embed, out, nullptr, 32000, 1024);
}
